// LSTM_84095459656154
// MI455X (gfx1250) — matrix operand order repaired and checked by reading, not compiled, not run on hardware
//
#include <hip/hip_runtime.h>

// ---------------------------------------------------------------------------
// LSTM on gfx1250 via bf16 WMMA (D = A x B + C, fp32 accumulate).
//   T=512, B=128, I=256, H=512, HX=768.
//   - Phase 1: x-projection for all T as one big WMMA GEMM, output in
//     tile-major layout (per-lane-contiguous accumulator fragments).
//   - Phase 2: 512 stream-ordered step kernels; recurrent GEMM h@Wh with
//     4-gate A reuse, double-buffered B tiles, fused gating.
//   - h handed between steps as bf16 already in WMMA-A lane layout (hpack,
//     parity double-buffered) -> next step's A staging is a raw 16KB copy.
// ---------------------------------------------------------------------------

typedef __attribute__((ext_vector_type(16))) __bf16 v16bf;
typedef __attribute__((ext_vector_type(8)))  float  v8f;

#define TT 512
#define BB 128
#define II 256
#define HH 512
#define G4 2048

__device__ __forceinline__ unsigned short f2bf(float f) {
  unsigned int u = __float_as_uint(f);
  unsigned int r = u + 0x7FFFu + ((u >> 16) & 1u);   // round-to-nearest-even
  return (unsigned short)(r >> 16);
}
__device__ __forceinline__ float sigf(float x)     { return 1.0f / (1.0f + __expf(-x)); }
__device__ __forceinline__ float tanhfast(float x) { return 1.0f - 2.0f / (__expf(2.0f * x) + 1.0f); }

// WMMA-A 16x32 bf16 lane layout (ISA 7.12.2):
//   lane = (m&15) + ((kk&15)>=8 ? 16 : 0);  elem e = (kk>=16)*8 + (kk&7)
__device__ __forceinline__ int a_lane(int m, int kk) { return (m & 15) + (((kk & 15) >= 8) ? 16 : 0); }
__device__ __forceinline__ int a_elem(int kk)        { return ((kk >= 16) ? 8 : 0) + (kk & 7); }

// WMMA-B 32x16 bf16 packed tiles: lanes 0-15: N=lane, K=(e>=8)*16+(e&7); lanes 16-31: N=lane-16, K=8+(e>=8)*16+(e&7)
// ws index: (((g*KT + kt)*32 + nt)*32 + lane)*16 + e
__device__ __forceinline__ void load_b4(const unsigned short* __restrict__ base, int KT,
                                        int kt, int ntH, int lane, v16bf* dst) {
#pragma unroll
  for (int g = 0; g < 4; ++g)
    dst[g] = *(const v16bf*)(base + (((g * KT + kt) * 32 + ntH) * 32 + lane) * 16);
}

// ============================================================================
// Weight packing (fp32 -> bf16 WMMA-B tile layout)
// ============================================================================
__global__ void pack_wh(const float* __restrict__ Wf, const float* __restrict__ Wi,
                        const float* __restrict__ Wc, const float* __restrict__ Wo,
                        unsigned short* __restrict__ bh) {
  int idx = blockIdx.x * 256 + threadIdx.x;          // 4*16*32*32*16 = 1048576
  int e = idx & 15, lane = (idx >> 4) & 31, nt = (idx >> 9) & 31;
  int kt = (idx >> 14) & 15, g = idx >> 18;
  const float* W = (g == 0) ? Wf : (g == 1) ? Wi : (g == 2) ? Wc : Wo;
  int kk = ((e >= 8) ? 16 : 0) + ((lane >= 16) ? 8 : 0) + (e & 7);
  int n = nt * 16 + (lane & 15);
  bh[idx] = f2bf(W[(kt * 32 + kk) * HH + n]);        // h-part rows 0..511
}

__global__ void pack_wx(const float* __restrict__ Wf, const float* __restrict__ Wi,
                        const float* __restrict__ Wc, const float* __restrict__ Wo,
                        unsigned short* __restrict__ bx) {
  int idx = blockIdx.x * 256 + threadIdx.x;          // 4*8*32*32*16 = 524288
  int e = idx & 15, lane = (idx >> 4) & 31, nt = (idx >> 9) & 31;
  int kt = (idx >> 14) & 7, g = idx >> 17;
  const float* W = (g == 0) ? Wf : (g == 1) ? Wi : (g == 2) ? Wc : Wo;
  int kk = ((e >= 8) ? 16 : 0) + ((lane >= 16) ? 8 : 0) + (e & 7);
  int n = nt * 16 + (lane & 15);
  bx[idx] = f2bf(W[(HH + kt * 32 + kk) * HH + n]);   // x-part rows 512..767
}

__global__ void zero_state(float* __restrict__ Cst, unsigned int* __restrict__ hp) {
  int i = blockIdx.x * 256 + threadIdx.x;            // 65536 threads
  Cst[i] = 0.0f;                                     // 128x512 fp32 C state
  hp[i] = 0u;                                        // both hpack buffers (256KB)
}

// ============================================================================
// Phase 1: gates_x = X @ Wx + bias ; GEMM [65536,256] @ [256,2048]
// gx tile-major: ((mTileGlobal*4 + g)*32 + ntH)*256 + lane*8  (fp32)
// ============================================================================
__global__ void __launch_bounds__(512)
lstm_xproj(const float* __restrict__ x, const unsigned short* __restrict__ bx,
           const float* __restrict__ bfv, const float* __restrict__ biv,
           const float* __restrict__ bcv, const float* __restrict__ bov,
           float* __restrict__ gx) {
  __shared__ unsigned short ldsA[8 * 512];           // 8 kTiles of 16x32 bf16
  const int tid = threadIdx.x;
  const int mTile = blockIdx.x;                      // 0..4095
  const float* xrow = x + (long)mTile * 16 * II;

  // Stage A slab (16x256): float4 loads, 4-wide bf16 pack, b64 LDS stores.
  for (int c = tid; c < 1024; c += 512) {
    int m = c >> 6;                                  // 0..15
    int k0 = (c & 63) * 4;                           // 0..252, multiple of 4
    float4 v = *(const float4*)(xrow + m * II + k0);
    int kt = k0 >> 5, kk = k0 & 31;
    int la = a_lane(m, kk), e0 = a_elem(kk);         // 4 elems land at e0..e0+3
    unsigned long long p = (unsigned long long)f2bf(v.x)
                         | ((unsigned long long)f2bf(v.y) << 16)
                         | ((unsigned long long)f2bf(v.z) << 32)
                         | ((unsigned long long)f2bf(v.w) << 48);
    *(unsigned long long*)&ldsA[kt * 512 + la * 16 + e0] = p;
  }
  __syncthreads();

  const int wave = tid >> 5, lane = tid & 31;
  const int nlo = lane & 15, hi = lane >> 4;

  for (int half = 0; half < 2; ++half) {
    const int ntH = wave + 16 * half;                // 0..31
    const int n = ntH * 16 + nlo;
    v8f acc[4];
    {
      float b0 = bfv[n], b1 = biv[n], b2 = bcv[n], b3 = bov[n];
#pragma unroll
      for (int r = 0; r < 8; ++r) { acc[0][r] = b0; acc[1][r] = b1; acc[2][r] = b2; acc[3][r] = b3; }
    }
    v16bf bb[2][4];
    load_b4(bx, 8, 0, ntH, lane, bb[0]);             // prologue
#pragma unroll
    for (int kt = 0; kt < 8; ++kt) {
      if (kt < 7) load_b4(bx, 8, kt + 1, ntH, lane, bb[(kt + 1) & 1]);
      v16bf a = *(const v16bf*)&ldsA[kt * 512 + lane * 16];
#pragma unroll
      for (int g = 0; g < 4; ++g)
        acc[g] = __builtin_amdgcn_wmma_f32_16x16x32_bf16(false, a, false, bb[kt & 1][g],
                                                         (short)0, acc[g], false, false);
    }
    long base = (((long)mTile * 4) * 32 + ntH) * 256 + lane * 8;
#pragma unroll
    for (int g = 0; g < 4; ++g)
      *(v8f*)(gx + base + (long)g * 8192) = acc[g];  // 32B contiguous per lane
  }
}

// ============================================================================
// Phase 2: one launch per t (stream order = recurrence sync).
// Grid 16 blocks x 512 thr. mTile = bid&7, ntGroup = bid>>3, wave owns ntH.
// hpack: bf16 h in WMMA-A layout, [2][mTile(8)][kt(16)][lane(32)][e(16)]
// ============================================================================
__global__ void __launch_bounds__(512)
lstm_step(int t, const unsigned short* __restrict__ bh, const float* __restrict__ gx,
          float* __restrict__ Cst, unsigned short* __restrict__ hpack,
          float* __restrict__ out) {
  __shared__ unsigned short ldsA[16 * 512];          // 16KB A slab
  const int tid = threadIdx.x, bid = blockIdx.x;
  const int mTile = bid & 7, ntGroup = bid >> 3;
  const unsigned short* hp_read = hpack + (t & 1) * 65536;
  unsigned short* hp_write = hpack + ((t + 1) & 1) * 65536;

  // Stage h_{t-1}: raw coalesced 16KB copy (already bf16, already A layout).
  {
    const uint4* src = (const uint4*)(hp_read + mTile * 8192);
    uint4* dst = (uint4*)ldsA;
    for (int i = tid; i < 1024; i += 512) dst[i] = src[i];
  }
  __syncthreads();

  const int wave = tid >> 5, lane = tid & 31;
  const int nlo = lane & 15, hi = lane >> 4;
  const int ntH = ntGroup * 16 + wave;               // 0..31
  const int n = ntH * 16 + nlo;

  // Accumulator init from tile-major gates_x: one 32B vector load per gate.
  v8f acc[4];
  {
    long base = ((((long)t * 8 + mTile) * 4) * 32 + ntH) * 256 + lane * 8;
#pragma unroll
    for (int g = 0; g < 4; ++g)
      acc[g] = *(const v8f*)(gx + base + (long)g * 8192);
  }

  // Recurrent GEMM, K=512: 16 kt steps x 4 gates, double-buffered B.
  v16bf bb[2][4];
  load_b4(bh, 16, 0, ntH, lane, bb[0]);
#pragma unroll
  for (int kt = 0; kt < 16; ++kt) {
    if (kt < 15) load_b4(bh, 16, kt + 1, ntH, lane, bb[(kt + 1) & 1]);
    v16bf a = *(const v16bf*)&ldsA[kt * 512 + lane * 16];
#pragma unroll
    for (int g = 0; g < 4; ++g)
      acc[g] = __builtin_amdgcn_wmma_f32_16x16x32_bf16(false, a, false, bb[kt & 1][g],
                                                       (short)0, acc[g], false, false);
  }

  // Fused gating + C update; emit h as fp32 (d_out) and bf16 A-layout (hpack).
  const int kt_w = ntH >> 1;                         // h column block this wave owns
  const int e_w = (ntH & 1) * 8 + (nlo & 7);
  const int laneHiSel = (nlo >= 8) ? 16 : 0;
#pragma unroll
  for (int r = 0; r < 8; ++r) {
    int mloc = r + hi * 8;                           // 0..15
    int row = mTile * 16 + mloc;                     // batch row 0..127
    float f  = sigf(acc[0][r]);
    float i_ = sigf(acc[1][r]);
    float c  = tanhfast(acc[2][r]);
    float o  = sigf(acc[3][r]);
    long ci = (long)row * HH + n;
    float Cn = f * Cst[ci] + i_ * c;
    Cst[ci] = Cn;
    float h = o * tanhfast(Cn);
    out[(long)t * (BB * HH) + ci] = h;
    hp_write[((mTile * 16 + kt_w) * 32 + (mloc + laneHiSel)) * 16 + e_w] = f2bf(h);
  }
}

// ============================================================================
// Workspace layout (bytes):
//   [0,       2MB)   : bh   packed recurrent weights, bf16
//   [2MB,     3MB)   : bx   packed input weights, bf16
//   [3MB,     +256K) : Cst  cell state fp32 [128,512]
//   [3.25MB,  +256K) : hpack bf16 h in A layout, x2 parity buffers
//   [4MB,     +512MB): gx   gates_x fp32, tile-major
// ============================================================================
extern "C" void kernel_launch(void* const* d_in, const int* in_sizes, int n_in,
                              void* d_out, int out_size, void* d_ws, size_t ws_size,
                              hipStream_t stream) {
  const float* x  = (const float*)d_in[0];
  const float* Wf = (const float*)d_in[1];
  const float* bf = (const float*)d_in[2];
  const float* Wi = (const float*)d_in[3];
  const float* bi = (const float*)d_in[4];
  const float* Wc = (const float*)d_in[5];
  const float* bc = (const float*)d_in[6];
  const float* Wo = (const float*)d_in[7];
  const float* bo = (const float*)d_in[8];
  float* out = (float*)d_out;

  char* ws = (char*)d_ws;
  unsigned short* bh    = (unsigned short*)(ws);
  unsigned short* bx    = (unsigned short*)(ws + (2u << 20));
  float*          Cst   = (float*)(ws + (3u << 20));
  unsigned short* hpack = (unsigned short*)(ws + (3u << 20) + (256u << 10));
  float*          gx    = (float*)(ws + (4u << 20));

  pack_wh<<<4096, 256, 0, stream>>>(Wf, Wi, Wc, Wo, bh);
  pack_wx<<<2048, 256, 0, stream>>>(Wf, Wi, Wc, Wo, bx);
  zero_state<<<256, 256, 0, stream>>>(Cst, (unsigned int*)hpack);
  lstm_xproj<<<4096, 512, 0, stream>>>(x, bx, bf, bi, bc, bo, gx);
  for (int t = 0; t < TT; ++t)
    lstm_step<<<16, 512, 0, stream>>>(t, bh, gx, Cst, hpack, out);
}
